// BatchGLRLoss_13786845020845
// MI455X (gfx1250) — compile-verified, matrix operand order checked by reading, not run on hardware
//
#include <hip/hip_runtime.h>
#include <hip/hip_bf16.h>

// MI455X / gfx1250: wave32, WMMA f32_16x16x32_f16 path.
//
// loss = (1/2n) * [ 2 * sum_i sum_{j in knn(i)} d2_ij  -  sum_{mutual (i,j)} d2_ij ]
//
// Pipeline:
//  k_prep       : z fp32 -> zh f16, sq[i] = ||z_i||^2 (exact fp32)
//  k_knn        : WMMA Gram sweep (double-buffered B frags, 4-way column split)
//                 + threshold-filtered per-row top-10 partials (wave-local merge)
//  k_mergeparts : merge the 4 partial top-10 lists per row
//  k_mutual     : mutuality check over the 8192x10 index lists, partial sums
//  k_final      : fixed-order reduction -> scalar loss

typedef _Float16 half_t;
typedef __attribute__((ext_vector_type(16))) _Float16 v16h;
typedef __attribute__((ext_vector_type(8)))  _Float16 v8h;
typedef __attribute__((ext_vector_type(4)))  _Float16 v4h;
typedef __attribute__((ext_vector_type(8)))  float    v8f;

#define BN     8192          // number of points
#define DDIM   512           // feature dim
#define KNN    10            // neighbors
#define CSPLIT 4             // column splits (partial top-K lists per row)
#define CRANGE (BN / CSPLIT) // columns per block = 2048
#define FBIG   3.402823466e+38f

// Compiler-ordering fence for wave-internal LDS communication.
// LDS ops from one wave are in-order in hardware (DScnt pipeline); we only
// need to stop the compiler from reordering across the phase boundary.
__device__ __forceinline__ void wave_lds_fence() {
  __builtin_amdgcn_fence(__ATOMIC_ACQ_REL, "wavefront");
  __builtin_amdgcn_wave_barrier();
}

// ---------------------------------------------------------------------------
// Kernel 1: fp32 -> f16 conversion + exact squared norms.
// ---------------------------------------------------------------------------
__global__ __launch_bounds__(256) void k_prep(const float* __restrict__ z,
                                              half_t* __restrict__ zh,
                                              float* __restrict__ sq) {
  const int wave = threadIdx.x >> 5;
  const int lane = threadIdx.x & 31;
  const int row  = blockIdx.x * 8 + wave;
  const float* zr  = z  + (size_t)row * DDIM;
  half_t*      zhr = zh + (size_t)row * DDIM;

  float acc = 0.0f;
#pragma unroll
  for (int t = 0; t < 4; ++t) {
    const int idx = t * 32 + lane;             // float4 index
    float4 v = ((const float4*)zr)[idx];
    acc = fmaf(v.x, v.x, acc);
    acc = fmaf(v.y, v.y, acc);
    acc = fmaf(v.z, v.z, acc);
    acc = fmaf(v.w, v.w, acc);
    v4h hv = { (_Float16)v.x, (_Float16)v.y, (_Float16)v.z, (_Float16)v.w };
    ((v4h*)zhr)[idx] = hv;
  }
#pragma unroll
  for (int off = 16; off > 0; off >>= 1)
    acc += __shfl_down(acc, off, 32);
  if (lane == 0) sq[row] = acc;
}

// ---------------------------------------------------------------------------
// Kernel 2: WMMA Gram sweep + per-row partial top-10.
// grid = (64 row-blocks, 4 col-splits), 256 threads (8 waves).
// Each wave owns 16 rows; A fragments (16 x v16h) live in registers for the
// whole sweep; B fragments are double-buffered in registers.
// ---------------------------------------------------------------------------
__global__ __launch_bounds__(256)
__attribute__((amdgpu_waves_per_eu(2)))
void k_knn(const half_t* __restrict__ zh,
           const float* __restrict__ sq,
           float* __restrict__ topv_p,     // [CSPLIT][BN][KNN]
           int* __restrict__ topi_p) {
  __shared__ float    topv[128][KNN];
  __shared__ int      topi[128][KNN];
  __shared__ float    thr[128];
  __shared__ unsigned qcnt[128];
  __shared__ float    qv[128][16];
  __shared__ int      qi[128][16];

  const int tid   = threadIdx.x;
  const int lane  = tid & 31;
  const int wave  = tid >> 5;
  const int m     = lane & 15;      // A-row / B-col within tile
  const int h     = lane >> 4;      // K-half selector (ISA 16-bit layout)
  const int rblk  = blockIdx.x * 128;
  const int rbase = rblk + wave * 16;
  const int cstart = blockIdx.y * CRANGE;
  const int cend   = cstart + CRANGE;

  // init shared state
  for (int x = tid; x < 128 * KNN; x += 256) {
    topv[x / KNN][x % KNN] = FBIG;
    topi[x / KNN][x % KNN] = 0;
  }
  for (int x = tid; x < 128; x += 256) { thr[x] = FBIG; qcnt[x] = 0u; }
  __syncthreads();

  // Load A fragments (rows rbase..rbase+15, all K) into registers.
  const half_t* arow = zh + (size_t)(rbase + m) * DDIM + 8 * h;
  v16h A[16];
#pragma unroll
  for (int kk = 0; kk < 16; ++kk) {
    v8h lo = *(const v8h*)(arow + 32 * kk);
    v8h hi = *(const v8h*)(arow + 32 * kk + 16);
    A[kk] = __builtin_shufflevector(lo, hi, 0,1,2,3,4,5,6,7,8,9,10,11,12,13,14,15);
  }

  // Per-lane exact squared norms of the 8 rows this lane accumulates
  // (C/D layout: lane holds N = lane%16, M = g + 8*(lane/16)).
  float sqr[8];
#pragma unroll
  for (int g = 0; g < 8; ++g) sqr[g] = sq[rbase + g + 8 * h];

  // B-tile loader: 16 fragments for columns c0..c0+15.
  auto load_btile = [&](v16h (&Bf)[16], int c0) {
    const half_t* brow = zh + (size_t)(c0 + m) * DDIM + 8 * h;
#pragma unroll
    for (int kk = 0; kk < 16; ++kk) {
      v8h lo = *(const v8h*)(brow + 32 * kk);
      v8h hi = *(const v8h*)(brow + 32 * kk + 16);
      Bf[kk] = __builtin_shufflevector(lo, hi, 0,1,2,3,4,5,6,7,8,9,10,11,12,13,14,15);
    }
  };

  // Process one 16x16 tile: WMMA accumulate, d2 filter, wave-local merge.
  auto process = [&](const v16h (&Bf)[16], int c0) {
    v8f acc = {};
#pragma unroll
    for (int kk = 0; kk < 16; ++kk) {
      acc = __builtin_amdgcn_wmma_f32_16x16x32_f16(
          false, A[kk], false, Bf[kk], (short)0, acc, false, false);
    }

    const int   j   = c0 + m;
    const float sqj = sq[j];
#pragma unroll
    for (int g = 0; g < 8; ++g) {
      const int rl = wave * 16 + g + 8 * h;   // row-local id (wave-exclusive)
      const int r  = rblk + rl;               // global row
      float d2 = sqr[g] + sqj - 2.0f * acc[g];
      if (r == j) d2 = FBIG;                  // exclude self
      if (d2 < thr[rl]) {
        unsigned p = atomicAdd(&qcnt[rl], 1u);  // ds_add_rtn_u32
        if (p < 16u) { qv[rl][p] = d2; qi[rl][p] = j; }
      }
    }
    wave_lds_fence();

    // Merge survivors: lanes 0..15 each own one of this wave's 16 rows.
    if (lane < 16) {
      const int rl = wave * 16 + lane;
      unsigned cnt = qcnt[rl];
      if (cnt > 16u) cnt = 16u;
      for (unsigned t = 0; t < cnt; ++t) {
        const float v  = qv[rl][t];
        const int   jj = qi[rl][t];
        if (v < topv[rl][KNN - 1]) {
          int p = KNN - 1;
          while (p > 0 && topv[rl][p - 1] > v) {
            topv[rl][p] = topv[rl][p - 1];
            topi[rl][p] = topi[rl][p - 1];
            --p;
          }
          topv[rl][p] = v;
          topi[rl][p] = jj;
        }
      }
      thr[rl]  = topv[rl][KNN - 1];
      qcnt[rl] = 0u;
    }
    wave_lds_fence();
  };

  // Double-buffered sweep over this block's column range (128 tiles).
  v16h B0[16], B1[16];
  load_btile(B0, cstart);
  for (int c0 = cstart; c0 < cend; c0 += 32) {
    load_btile(B1, c0 + 16);
    process(B0, c0);
    if (c0 + 32 < cend) load_btile(B0, c0 + 32);
    process(B1, c0 + 16);
  }

  __syncthreads();

  // Emit per-row partial results for this column split.
  if (tid < 128) {
    const int r = rblk + tid;
    const size_t base = ((size_t)blockIdx.y * BN + r) * KNN;
#pragma unroll
    for (int t = 0; t < KNN; ++t) {
      topv_p[base + t] = topv[tid][t];
      topi_p[base + t] = topi[tid][t];
    }
  }
}

// ---------------------------------------------------------------------------
// Kernel 3: merge the CSPLIT partial top-10 lists per row (LDS insertion).
// 32 blocks x 256 threads = 8192 rows.
// ---------------------------------------------------------------------------
__global__ __launch_bounds__(256) void k_mergeparts(const float* __restrict__ topv_p,
                                                    const int* __restrict__ topi_p,
                                                    float* __restrict__ topv_g,
                                                    int* __restrict__ topi_g) {
  __shared__ float mv[256][KNN];
  __shared__ int   mi[256][KNN];
  const int t0 = threadIdx.x;
  const int i  = blockIdx.x * 256 + t0;

#pragma unroll
  for (int t = 0; t < KNN; ++t) { mv[t0][t] = FBIG; mi[t0][t] = 0; }

  for (int s = 0; s < CSPLIT; ++s) {
    const size_t base = ((size_t)s * BN + i) * KNN;
#pragma unroll
    for (int t = 0; t < KNN; ++t) {
      const float v  = topv_p[base + t];
      const int   jj = topi_p[base + t];
      if (v < mv[t0][KNN - 1]) {
        int p = KNN - 1;
        while (p > 0 && mv[t0][p - 1] > v) {
          mv[t0][p] = mv[t0][p - 1];
          mi[t0][p] = mi[t0][p - 1];
          --p;
        }
        mv[t0][p] = v;
        mi[t0][p] = jj;
      }
    }
  }
#pragma unroll
  for (int t = 0; t < KNN; ++t) {
    topv_g[i * KNN + t] = mv[t0][t];
    topi_g[i * KNN + t] = mi[t0][t];
  }
}

// ---------------------------------------------------------------------------
// Kernel 4: mutuality pass. edge (i,j in knn(i)) contributes d2 * (mutual?1:2).
// ---------------------------------------------------------------------------
__global__ __launch_bounds__(256) void k_mutual(const int* __restrict__ topi_g,
                                                const float* __restrict__ topv_g,
                                                float* __restrict__ partial) {
  const int i = blockIdx.x * 256 + threadIdx.x;
  float s = 0.0f;
#pragma unroll
  for (int t = 0; t < KNN; ++t) {
    const int   j = topi_g[i * KNN + t];
    const float v = topv_g[i * KNN + t];
    bool mut = false;
#pragma unroll
    for (int u = 0; u < KNN; ++u) mut |= (topi_g[j * KNN + u] == i);
    s += mut ? v : 2.0f * v;
  }
  __shared__ float red[256];
  red[threadIdx.x] = s;
  __syncthreads();
  for (int off = 128; off > 0; off >>= 1) {
    if (threadIdx.x < off) red[threadIdx.x] += red[threadIdx.x + off];
    __syncthreads();
  }
  if (threadIdx.x == 0) partial[blockIdx.x] = red[0];
}

// ---------------------------------------------------------------------------
// Kernel 5: fixed-order final reduction -> loss scalar.
// ---------------------------------------------------------------------------
__global__ void k_final(const float* __restrict__ partial, float* __restrict__ out) {
  if (threadIdx.x == 0 && blockIdx.x == 0) {
    float s = 0.0f;
    for (int b = 0; b < 32; ++b) s += partial[b];
    out[0] = s / (2.0f * (float)BN);
  }
}

// ---------------------------------------------------------------------------
extern "C" void kernel_launch(void* const* d_in, const int* in_sizes, int n_in,
                              void* d_out, int out_size, void* d_ws, size_t ws_size,
                              hipStream_t stream) {
  (void)in_sizes; (void)n_in; (void)out_size; (void)ws_size;
  const float* z = (const float*)d_in[0];

  // Workspace layout (~11.7 MB total)
  char* ws = (char*)d_ws;
  const size_t off_zh   = 0;                                  // 8192*512*2  = 8388608
  const size_t off_sq   = off_zh   + (size_t)BN * DDIM * 2;   // 8192*4      = 32768
  const size_t off_tvp  = off_sq   + (size_t)BN * 4;          // 4*8192*10*4 = 1310720
  const size_t off_tip  = off_tvp  + (size_t)CSPLIT * BN * KNN * 4;
  const size_t off_tvg  = off_tip  + (size_t)CSPLIT * BN * KNN * 4; // 327680
  const size_t off_tig  = off_tvg  + (size_t)BN * KNN * 4;          // 327680
  const size_t off_part = off_tig  + (size_t)BN * KNN * 4;          // 128

  half_t* zh      = (half_t*)(ws + off_zh);
  float*  sq      = (float*)(ws + off_sq);
  float*  topv_p  = (float*)(ws + off_tvp);
  int*    topi_p  = (int*)  (ws + off_tip);
  float*  topv_g  = (float*)(ws + off_tvg);
  int*    topi_g  = (int*)  (ws + off_tig);
  float*  partial = (float*)(ws + off_part);

  k_prep      <<<BN / 8, 256, 0, stream>>>(z, zh, sq);
  k_knn       <<<dim3(BN / 128, CSPLIT), 256, 0, stream>>>(zh, sq, topv_p, topi_p);
  k_mergeparts<<<BN / 256, 256, 0, stream>>>(topv_p, topi_p, topv_g, topi_g);
  k_mutual    <<<BN / 256, 256, 0, stream>>>(topi_g, topv_g, partial);
  k_final     <<<1, 32, 0, stream>>>(partial, (float*)d_out);
}
